// DynamiConv_91156385890312
// MI455X (gfx1250) — matrix operand order repaired and checked by reading, not compiled, not run on hardware
//
#include <hip/hip_runtime.h>

typedef __attribute__((ext_vector_type(16))) _Float16 v16h;
typedef __attribute__((ext_vector_type(8)))  _Float16 v8h;
typedef __attribute__((ext_vector_type(8)))  float    v8f;

#define CCH   32            // input channels
#define TILE  32            // 32x32 output positions per workgroup
#define PADW  36            // padded x-stride (positions) of LDS tile
#define XT_HALFS (34*PADW*CCH)   // 39168 halfs per input tile
#define AH_HALFS (32*288)        // main-conv weights, A layout
#define DAH_HALFS (16*288)       // gate weights (row0 real, rows1-15 zero)
#define SMEM_BYTES ((2*XT_HALFS + AH_HALFS + DAH_HALFS) * 2)  // 184320

union V16 { v16h v; v8h h[2]; };

__global__ __launch_bounds__(256)
void dynconv_wmma_kernel(const float* __restrict__ xg,
                         const float* __restrict__ wg,
                         const float* __restrict__ dwg,
                         const float* __restrict__ bg,
                         float* __restrict__ outg)
{
    extern __shared__ _Float16 smem[];
    _Float16* xt  = smem;                 // f16 input tile  [34][36][32]
    _Float16* st  = xt + XT_HALFS;        // f16 sigmoid tile
    _Float16* Ah  = st + XT_HALFS;        // [32][288]  K = tap*32 + c
    _Float16* dAh = Ah + AH_HALFS;        // [16][288]  row0 = dweight

    const int tid = threadIdx.x;
    const int b   = blockIdx.x >> 6;
    const int t   = blockIdx.x & 63;
    const int y0  = (t >> 3) * TILE;
    const int x0  = (t & 7)  * TILE;

    // ---- Phase 0: global -> LDS, f32 -> f16, sigmoid once per element ----
    const float* xb = xg + (size_t)b * CCH * 256 * 256;
    for (int fi = tid; fi < CCH*34*34; fi += 256) {
        int c  = fi / (34*34);
        int rr = fi - c*(34*34);
        int yy = rr / 34;
        int xx = rr - yy*34;
        int y = y0 + yy - 1, x = x0 + xx - 1;
        float v = 0.f;
        if ((unsigned)y < 256u && (unsigned)x < 256u)
            v = xb[(size_t)c*65536 + y*256 + x];
        float s = 1.f / (1.f + __expf(-v));   // sigmoid(0)=0.5 for padding: matches ref
        int la = (yy*PADW + xx)*CCH + c;
        xt[la] = (_Float16)v;
        st[la] = (_Float16)s;
    }
    // weights: reorder [O][C][9] -> [O][tap*32+c], convert to f16
    for (int fi = tid; fi < AH_HALFS; fi += 256) {
        int o = fi / 288, k = fi - o*288;
        int r = k >> 5, c = k & 31;
        Ah[fi] = (_Float16)wg[(o*32 + c)*9 + r];
    }
    for (int fi = tid; fi < DAH_HALFS; fi += 256) {
        int m = fi / 288, k = fi - m*288;
        int r = k >> 5, c = k & 31;
        dAh[fi] = (m == 0) ? (_Float16)dwg[c*9 + r] : (_Float16)0.f;
    }
    __syncthreads();

    const int lane = tid & 31;
    const int wv   = tid >> 5;       // wave 0..7
    const int lm   = lane & 15;
    const int lh   = lane >> 4;      // lane half selects K-half of fragments

    // Each wave owns 4 output rows; per row, two 16-wide n-tiles processed
    // together so A fragments amortize over 2 WMMAs each.
    #pragma unroll 1
    for (int q = 0; q < 4; ++q) {
        int row = wv*4 + q;          // output row in tile (0..31)
        v8f acc0a = {}, acc1a = {}, accga = {};   // n-tile at xoff=0
        v8f acc0b = {}, acc1b = {}, accgb = {};   // n-tile at xoff=16
        #pragma unroll
        for (int r = 0; r < 9; ++r) {
            int ki = r / 3, kj = r - ki*3;
            // A fragments (16-bit A layout: lanes0-15 K{0..7,16..23}, lanes16-31 K{8..15,24..31})
            V16 a0, a1, ag;
            const _Float16* ab = Ah + lm*288 + r*32 + lh*8;
            a0.h[0] = *(const v8h*)(ab);         a0.h[1] = *(const v8h*)(ab + 16);
            const _Float16* a1b = ab + 16*288;
            a1.h[0] = *(const v8h*)(a1b);        a1.h[1] = *(const v8h*)(a1b + 16);
            const _Float16* agb = dAh + lm*288 + r*32 + lh*8;
            ag.h[0] = *(const v8h*)(agb);        ag.h[1] = *(const v8h*)(agb + 16);
            // B fragments: 32ch x 16pos slabs at tap (ki,kj); lane n=lm, K-half = lh
            int la = ((row + ki)*PADW + lm + kj)*CCH + lh*8;       // xoff = 0
            int lb = la + 16*CCH;                                  // xoff = 16
            V16 bx0, bs0, bx1, bs1;
            bx0.h[0] = *(const v8h*)(xt + la);   bx0.h[1] = *(const v8h*)(xt + la + 16);
            bs0.h[0] = *(const v8h*)(st + la);   bs0.h[1] = *(const v8h*)(st + la + 16);
            bx1.h[0] = *(const v8h*)(xt + lb);   bx1.h[1] = *(const v8h*)(xt + lb + 16);
            bs1.h[0] = *(const v8h*)(st + lb);   bs1.h[1] = *(const v8h*)(st + lb + 16);

            acc0a = __builtin_amdgcn_wmma_f32_16x16x32_f16(false, a0.v, false, bx0.v, (short)0, acc0a, false, false);
            acc0b = __builtin_amdgcn_wmma_f32_16x16x32_f16(false, a0.v, false, bx1.v, (short)0, acc0b, false, false);
            acc1a = __builtin_amdgcn_wmma_f32_16x16x32_f16(false, a1.v, false, bx0.v, (short)0, acc1a, false, false);
            acc1b = __builtin_amdgcn_wmma_f32_16x16x32_f16(false, a1.v, false, bx1.v, (short)0, acc1b, false, false);
            accga = __builtin_amdgcn_wmma_f32_16x16x32_f16(false, ag.v, false, bs0.v, (short)0, accga, false, false);
            accgb = __builtin_amdgcn_wmma_f32_16x16x32_f16(false, ag.v, false, bs1.v, (short)0, accgb, false, false);
        }
        // gate row (M=0) lives in lanes 0-15, VGPR0; broadcast per column
        float g0 = __shfl(accga[0], lm, 32);
        float g1 = __shfl(accgb[0], lm, 32);
        int yg = y0 + row;
        size_t obase0 = ((size_t)b*32)*65536 + (size_t)yg*256 + (x0 + lm);
        size_t obase1 = obase0 + 16;
        #pragma unroll
        for (int v = 0; v < 8; ++v) {
            int o = v + lh*8;                      // D layout: lanes16-31 hold M=v+8
            float bo  = bg[o];
            float bo2 = bg[o+16];
            outg[obase0 + (size_t)o*65536]      = acc0a[v]*g0 + bo;
            outg[obase1 + (size_t)o*65536]      = acc0b[v]*g1 + bo;
            outg[obase0 + (size_t)(o+16)*65536] = acc1a[v]*g0 + bo2;
            outg[obase1 + (size_t)(o+16)*65536] = acc1b[v]*g1 + bo2;
        }
    }
}

extern "C" void kernel_launch(void* const* d_in, const int* in_sizes, int n_in,
                              void* d_out, int out_size, void* d_ws, size_t ws_size,
                              hipStream_t stream) {
    (void)in_sizes; (void)n_in; (void)out_size; (void)d_ws; (void)ws_size;
    const float* x    = (const float*)d_in[0];
    const float* w    = (const float*)d_in[1];
    const float* dw   = (const float*)d_in[2];
    const float* bias = (const float*)d_in[3];
    float* out = (float*)d_out;

    (void)hipFuncSetAttribute((const void*)&dynconv_wmma_kernel,
                              hipFuncAttributeMaxDynamicSharedMemorySize,
                              (int)SMEM_BYTES);
    dim3 grid(8 * 64);   // 8 batches x 64 spatial tiles (32x32 each)
    dim3 block(256);     // 8 waves of 32
    hipLaunchKernelGGL(dynconv_wmma_kernel, grid, block, SMEM_BYTES, stream,
                       x, w, dw, bias, out);
}
